// NodeClsHead_24180665876740
// MI455X (gfx1250) — matrix operand order repaired and checked by reading, not compiled, not run on hardware
//
#include <hip/hip_runtime.h>
#include <hip/hip_bf16.h>
#include <math.h>

#define NN 100000
#define NE 1600000
#define EPSF 1e-7f

typedef __attribute__((ext_vector_type(16))) __bf16 v16bf;
typedef __attribute__((ext_vector_type(8)))  __bf16 v8bf;
typedef __attribute__((ext_vector_type(8)))  float  v8f;

// ---------------- utility fills ----------------
__global__ void fill_f32(float* __restrict__ p, float v, long n) {
    long i = (long)blockIdx.x * blockDim.x + threadIdx.x;
    if (i < n) p[i] = v;
}

// ---------------- weight transpose+convert ----------------
// W1: 256x256 (k-major rows) -> W1T[n*256+k] bf16
__global__ void w1t_kernel(const float* __restrict__ W, __bf16* __restrict__ WT) {
    int idx = blockIdx.x * blockDim.x + threadIdx.x;   // 65536
    int n = idx >> 8, k = idx & 255;
    WT[idx] = (__bf16)W[k * 256 + n];
}
// W2: 256x40 -> W2T[48][256] bf16, rows 40..47 zero
__global__ void w2t_kernel(const float* __restrict__ W, __bf16* __restrict__ WT) {
    int idx = blockIdx.x * blockDim.x + threadIdx.x;   // 48*256
    int n = idx >> 8, k = idx & 255;
    WT[idx] = (n < 40) ? (__bf16)W[k * 40 + n] : (__bf16)0.0f;
}

// ---------------- feature construction (1 wave per node) ----------------
__global__ void feat_kernel(const float* __restrict__ x,
                            const float* __restrict__ xH,
                            const float* __restrict__ xS,
                            __bf16* __restrict__ feat) {
    int wave = threadIdx.x >> 5;
    int lane = threadIdx.x & 31;
    int node = blockIdx.x * 8 + wave;
    if (node >= NN) return;
    __bf16* fp = feat + (size_t)node * 256;

    // euclidean block: cols 0..127
    float4 xv = *(const float4*)(x + (size_t)node * 128 + lane * 4);
    fp[lane * 4 + 0] = (__bf16)xv.x;
    fp[lane * 4 + 1] = (__bf16)xv.y;
    fp[lane * 4 + 2] = (__bf16)xv.z;
    fp[lane * 4 + 3] = (__bf16)xv.w;

    int j0 = 1 + 2 * lane;
    int j1 = 2 + 2 * lane;

    // lorentz logmap0: cols 128..191
    {
        const float* H = xH + (size_t)node * 64;
        float a = H[j0];
        float b = (j1 < 64) ? H[j1] : 0.0f;
        float ss = a * a + b * b;
        for (int m = 16; m >= 1; m >>= 1) ss += __shfl_xor(ss, m, 32);
        float yn = fmaxf(sqrtf(ss), EPSF);
        float scale = acoshf(fmaxf(H[0], 1.0f + EPSF)) / yn;
        if (lane == 0) fp[128] = (__bf16)0.0f;
        fp[128 + j0] = (__bf16)(a * scale);
        if (j1 < 64) fp[128 + j1] = (__bf16)(b * scale);
    }
    // sphere logmap0: cols 192..255
    {
        const float* S = xS + (size_t)node * 64;
        float a = S[j0];
        float b = (j1 < 64) ? S[j1] : 0.0f;
        float ss = a * a + b * b;
        for (int m = 16; m >= 1; m >>= 1) ss += __shfl_xor(ss, m, 32);
        float yn = fmaxf(sqrtf(ss), EPSF);
        float c0 = fminf(fmaxf(S[0], -1.0f + EPSF), 1.0f - EPSF);
        float scale = acosf(c0) / yn;
        if (lane == 0) fp[192] = (__bf16)0.0f;
        fp[192 + j0] = (__bf16)(a * scale);
        if (j1 < 64) fp[192 + j1] = (__bf16)(b * scale);
    }
}

// ---------------- degree / dinv ----------------
__global__ void deg_kernel(const int* __restrict__ dst, float* __restrict__ deg, int e) {
    int i = blockIdx.x * blockDim.x + threadIdx.x;
    if (i < e) atomicAdd(&deg[dst[i]], 1.0f);
}
__global__ void dinv_kernel(float* __restrict__ d, int n) {
    int i = blockIdx.x * blockDim.x + threadIdx.x;
    if (i < n) d[i] = rsqrtf(d[i]);
}

// ---------------- WMMA GEMM: C[M x nmax] = A[M x 256] * B(T stored: Nc x 256) ----------------
// one wave per 16x16 output tile; K = 256, bf16 inputs, f32 accum
__global__ void gemm_wmma_bf16(const __bf16* __restrict__ A,
                               const __bf16* __restrict__ BT,
                               float* __restrict__ C,
                               int nTilesN, int numTiles, int ldc, int nmax) {
    const int K = 256;
    int lane = threadIdx.x & 31;
    int tile = blockIdx.x * (blockDim.x >> 5) + (threadIdx.x >> 5);
    if (tile >= numTiles) return;               // uniform per-wave: EXEC stays full
    int mt = tile / nTilesN;
    int nt = tile - mt * nTilesN;
    int l16 = lane & 15;
    int half = lane >> 4;

    // A frag (16-bit 16x32 layout): lane row = l16; K chunks at kk+8*half and kk+16+8*half
    const __bf16* arow = A + (size_t)(mt * 16 + l16) * K + 8 * half;
    // B frag (16-bit 32x16 layout): lane col = l16; K chunks at kk+8*half and kk+16+8*half
    const __bf16* brow = BT + (size_t)(nt * 16 + l16) * K + 8 * half;

    v8f acc = {};
#pragma unroll
    for (int kk = 0; kk < K; kk += 32) {
        v8bf a0 = *(const v8bf*)(arow + kk);
        v8bf a1 = *(const v8bf*)(arow + kk + 16);
        v16bf a = __builtin_shufflevector(a0, a1, 0, 1, 2, 3, 4, 5, 6, 7,
                                                  8, 9, 10, 11, 12, 13, 14, 15);
        v8bf b0 = *(const v8bf*)(brow + kk);
        v8bf b1 = *(const v8bf*)(brow + kk + 16);
        v16bf b = __builtin_shufflevector(b0, b1, 0, 1, 2, 3, 4, 5, 6, 7,
                                                  8, 9, 10, 11, 12, 13, 14, 15);
        acc = __builtin_amdgcn_wmma_f32_16x16x32_bf16(
            false, a, false, b, (short)0, acc, false, false);
    }

    int col = nt * 16 + l16;
    if (col < nmax) {
        float* crow = C + (size_t)(mt * 16 + 8 * half) * ldc + col;
#pragma unroll
        for (int j = 0; j < 8; ++j) crow[(size_t)j * ldc] = acc[j];
    }
}

// ---------------- edge scatter: agg[dst] += h[src] * dinv[src]*dinv[dst] ----------------
__global__ void scatter_kernel(const float* __restrict__ h, const int* __restrict__ src,
                               const int* __restrict__ dst, const float* __restrict__ dinv,
                               float* __restrict__ agg, int F, int chunks, long total) {
    long gid = (long)blockIdx.x * blockDim.x + threadIdx.x;
    if (gid >= total) return;
    int e = (int)(gid / chunks);
    int c = (int)(gid - (long)e * chunks);
    int s = src[e], d = dst[e];
    float coef = dinv[s] * dinv[d];
    float4 v = *(const float4*)(h + (size_t)s * F + c * 4);
    float* ap = agg + (size_t)d * F + c * 4;
    atomicAdd(ap + 0, v.x * coef);
    atomicAdd(ap + 1, v.y * coef);
    atomicAdd(ap + 2, v.z * coef);
    atomicAdd(ap + 3, v.w * coef);
}

// ---------------- combine stages ----------------
__global__ void combine1_kernel(const float* __restrict__ agg, const float* __restrict__ h,
                                const float* __restrict__ dinv, const float* __restrict__ b,
                                __bf16* __restrict__ out) {
    long gid = (long)blockIdx.x * blockDim.x + threadIdx.x;
    if (gid >= (long)NN * 256) return;
    int i = (int)(gid >> 8);
    int f = (int)(gid & 255);
    float di = dinv[i];
    float v = agg[gid] + h[gid] * di * di + b[f];
    out[gid] = (__bf16)fmaxf(v, 0.0f);
}
__global__ void combine2_kernel(const float* __restrict__ agg, const float* __restrict__ h,
                                const float* __restrict__ dinv, const float* __restrict__ b,
                                float* __restrict__ out) {
    long gid = (long)blockIdx.x * blockDim.x + threadIdx.x;
    if (gid >= (long)NN * 40) return;
    int i = (int)(gid / 40);
    int f = (int)(gid - (long)i * 40);
    float di = dinv[i];
    out[gid] = agg[gid] + h[gid] * di * di + b[f];
}

extern "C" void kernel_launch(void* const* d_in, const int* in_sizes, int n_in,
                              void* d_out, int out_size, void* d_ws, size_t ws_size,
                              hipStream_t stream) {
    (void)in_sizes; (void)n_in; (void)out_size; (void)ws_size;
    const float* x   = (const float*)d_in[0];
    const float* xH  = (const float*)d_in[1];
    const float* xS  = (const float*)d_in[2];
    const int*   src = (const int*)d_in[3];
    const int*   dst = (const int*)d_in[4];
    const float* W1  = (const float*)d_in[5];
    const float* b1  = (const float*)d_in[6];
    const float* W2  = (const float*)d_in[7];
    const float* b2  = (const float*)d_in[8];
    float* out = (float*)d_out;

    char* base = (char*)d_ws;
    size_t off = 0;
    auto alloc = [&](size_t bytes) -> void* {
        off = (off + 255) & ~(size_t)255;
        void* p = base + off;
        off += bytes;
        return p;
    };
    float*  dinv = (float*)alloc((size_t)NN * 4);              // deg then dinv in place
    __bf16* feat = (__bf16*)alloc((size_t)NN * 256 * 2);       // also reused as relu-out bf16
    __bf16* w1t  = (__bf16*)alloc((size_t)256 * 256 * 2);
    __bf16* w2t  = (__bf16*)alloc((size_t)48 * 256 * 2);
    float*  h1   = (float*)alloc((size_t)NN * 256 * 4);
    float*  agg1 = (float*)alloc((size_t)NN * 256 * 4);
    float*  h2   = (float*)alloc((size_t)NN * 40 * 4);
    float*  agg2 = (float*)alloc((size_t)NN * 40 * 4);

    const int TB = 256;
    // init
    fill_f32<<<(NN + TB - 1) / TB, TB, 0, stream>>>(dinv, 1.0f, (long)NN);
    fill_f32<<<(int)(((long)NN * 256 + TB - 1) / TB), TB, 0, stream>>>(agg1, 0.0f, (long)NN * 256);
    fill_f32<<<(int)(((long)NN * 40 + TB - 1) / TB), TB, 0, stream>>>(agg2, 0.0f, (long)NN * 40);
    // weights
    w1t_kernel<<<256 * 256 / TB, TB, 0, stream>>>(W1, w1t);
    w2t_kernel<<<48 * 256 / TB, TB, 0, stream>>>(W2, w2t);
    // features (1 wave/node, 8 nodes per block)
    feat_kernel<<<NN / 8, TB, 0, stream>>>(x, xH, xS, feat);
    // degrees
    deg_kernel<<<NE / TB, TB, 0, stream>>>(dst, dinv, NE);
    dinv_kernel<<<(NN + TB - 1) / TB, TB, 0, stream>>>(dinv, NN);
    // layer 1: h1 = feat @ W1   (M tiles 6250, N tiles 16 -> 100000 tiles, 8 waves/block)
    {
        int numTiles = (NN / 16) * 16;
        gemm_wmma_bf16<<<(numTiles + 7) / 8, TB, 0, stream>>>(feat, w1t, h1, 16, numTiles, 256, 256);
    }
    // layer 1 message passing
    {
        long total = (long)NE * 64;
        scatter_kernel<<<(int)((total + TB - 1) / TB), TB, 0, stream>>>(h1, src, dst, dinv, agg1, 256, 64, total);
    }
    // relu combine -> bf16 (reuse feat buffer)
    combine1_kernel<<<(int)(((long)NN * 256 + TB - 1) / TB), TB, 0, stream>>>(agg1, h1, dinv, b1, feat);
    // layer 2: h2 = relu_out @ W2  (N tiles 3, cols < 40 stored, ldc = 40)
    {
        int numTiles = (NN / 16) * 3;
        gemm_wmma_bf16<<<(numTiles + 7) / 8, TB, 0, stream>>>(feat, w2t, h2, 3, numTiles, 40, 40);
    }
    // layer 2 message passing
    {
        long total = (long)NE * 10;
        scatter_kernel<<<(int)((total + TB - 1) / TB), TB, 0, stream>>>(h2, src, dst, dinv, agg2, 40, 10, total);
    }
    // final combine -> output
    combine2_kernel<<<(int)(((long)NN * 40 + TB - 1) / TB), TB, 0, stream>>>(agg2, h2, dinv, b2, out);
}
